// ESGRETLayer_71451075936283
// MI455X (gfx1250) — matrix operand order repaired and checked by reading, not compiled, not run on hardware
//
#include <hip/hip_runtime.h>

#define N_NODES 50000
#define N_EDGES 800000
#define D_IN    128
#define D_OUT   64
#define NEG_SLOPE 0.01f

typedef __attribute__((ext_vector_type(16))) _Float16 v16h;
typedef __attribute__((ext_vector_type(8)))  float    v8f;

// ---- workspace layout (32-bit word offsets) ----
#define WS_Z     0                          // N_NODES*D_OUT floats
#define WS_ASRC  (WS_Z    + N_NODES*D_OUT)  // N_NODES floats
#define WS_ADST  (WS_ASRC + N_NODES)        // N_NODES floats
#define WS_MENC  (WS_ADST + N_NODES)        // N_NODES uints (encoded segment max)
#define WS_S     (WS_MENC + N_NODES)        // N_NODES floats (segment sum)
#define WS_V     (WS_S    + N_NODES)        // 2 floats: W_edge.T @ attn_edge
#define WS_EX    (WS_V    + 64)             // N_EDGES*2 floats
#define WS_LOGIT (WS_EX   + N_EDGES*2)      // N_EDGES floats

// monotone float<->uint order-preserving encoding for atomicMax on floats
__device__ __forceinline__ unsigned fenc(float f) {
    unsigned u = __float_as_uint(f);
    return (u & 0x80000000u) ? ~u : (u | 0x80000000u);
}
__device__ __forceinline__ float fdec(unsigned u) {
    return __uint_as_float((u & 0x80000000u) ? (u ^ 0x80000000u) : ~u);
}

// ---------------- Kernel 1: z = h @ W_fc.T via WMMA f16 ----------------
// One wave per 16-row M tile; 4 N-tiles of 16 => whole 16x64 output block.
__global__ void gemm_z_kernel(const float* __restrict__ h,
                              const float* __restrict__ Wfc,
                              float* __restrict__ z) {
    const int wavesPerBlock = blockDim.x >> 5;
    const int wave = blockIdx.x * wavesPerBlock + (threadIdx.x >> 5);
    if (wave >= N_NODES / 16) return;                 // uniform per wave: EXEC stays all-1s
    const int lane   = threadIdx.x & 31;
    const int l15    = lane & 15;
    const int hiHalf = lane >> 4;                     // 0 or 1
    const int m      = wave * 16 + l15;               // row owned by this lane (A frag)

    v8f acc[4] = {};                                  // 4 N-tiles of 16x16 f32

    #pragma unroll
    for (int kk = 0; kk < D_IN; kk += 32) {
        // A fragment: 16x32 f16. Lane layout: elems 0-15 -> K=ka..ka+15
        const int ka = kk + hiHalf * 16;
        const float* __restrict__ hp = h + (size_t)m * D_IN + ka;
        v16h afrag;
        #pragma unroll
        for (int j = 0; j < 8; ++j) afrag[j]     = (_Float16)hp[j];
        #pragma unroll
        for (int j = 0; j < 8; ++j) afrag[8 + j] = (_Float16)hp[8 + j];

        #pragma unroll
        for (int t = 0; t < 4; ++t) {
            // B fragment: 32x16 f16. Lane = column n; low lanes K=kk..kk+15, high lanes K=kk+16..kk+31.
            // B[k][n] = W_fc[n][k] (row n of W_fc is contiguous over K).
            const int n  = t * 16 + l15;
            const int kb = kk + hiHalf * 16;
            const float* __restrict__ wp = Wfc + (size_t)n * D_IN + kb;
            v16h bfrag;
            #pragma unroll
            for (int j = 0; j < 16; ++j) bfrag[j] = (_Float16)wp[j];

            acc[t] = __builtin_amdgcn_wmma_f32_16x16x32_f16(
                false, afrag, false, bfrag, (short)0, acc[t], false, false);
        }
    }

    // C/D layout: VGPR j holds (M = base + hiHalf*8 + j, N = t*16 + l15)
    #pragma unroll
    for (int t = 0; t < 4; ++t) {
        const int n = t * 16 + l15;
        #pragma unroll
        for (int j = 0; j < 8; ++j) {
            const int row = wave * 16 + hiHalf * 8 + j;
            z[(size_t)row * D_OUT + n] = acc[t][j];
        }
    }
}

// ---------------- Kernel 2: v = W_edge.T @ attn_edge (2 scalars) ----------------
__global__ void param_kernel(const float* __restrict__ Wedge,
                             const float* __restrict__ attn_edge,
                             float* __restrict__ v) {
    const int t = threadIdx.x;
    if (t < 2) {
        float acc = 0.f;
        for (int d = 0; d < D_OUT; ++d) acc += Wedge[d * 2 + t] * attn_edge[d];
        v[t] = acc;
    }
}

// ---------------- Kernel 3: per-node attention scalars ----------------
// alpha_src[n] = z[n] . W_attn[0:64]; alpha_dst[n] = z[n] . W_attn[64:128]
__global__ void alpha_kernel(const float* __restrict__ z,
                             const float* __restrict__ Wattn,
                             float* __restrict__ asrc,
                             float* __restrict__ adst) {
    const int node = blockIdx.x * (blockDim.x >> 5) + (threadIdx.x >> 5);
    if (node >= N_NODES) return;
    const int lane = threadIdx.x & 31;
    const float* __restrict__ zr = z + (size_t)node * D_OUT;
    float z0 = zr[lane], z1 = zr[lane + 32];
    float pa = z0 * Wattn[lane]      + z1 * Wattn[lane + 32];
    float pb = z0 * Wattn[64 + lane] + z1 * Wattn[96 + lane];
    #pragma unroll
    for (int off = 16; off; off >>= 1) {
        pa += __shfl_xor(pa, off, 32);
        pb += __shfl_xor(pb, off, 32);
    }
    if (lane == 0) { asrc[node] = pa; adst[node] = pb; }
}

// ---------------- Kernel 4: edge pass 1 (logits + segment max) ----------------
__global__ void edge1_kernel(const float* __restrict__ e,
                             const int* __restrict__ src,
                             const int* __restrict__ dst,
                             const float* __restrict__ Weatt,
                             const float* __restrict__ Wattn,
                             const float* __restrict__ v,
                             const float* __restrict__ asrc,
                             const float* __restrict__ adst,
                             float* __restrict__ exw,
                             float* __restrict__ logitw,
                             unsigned* __restrict__ menc) {
    const int i = blockIdx.x * blockDim.x + threadIdx.x;
    if (i >= N_EDGES) return;
    const float e0 = e[2 * i], e1 = e[2 * i + 1];
    const float ex0 = e0 * Weatt[0] + e1 * Weatt[1];   // ex = e @ W_eatt.T
    const float ex1 = e0 * Weatt[2] + e1 * Weatt[3];
    // a = alpha_src[src] + alpha_dst[dst] + ex.(wc + v)    (v folds in the ee term)
    const float a = asrc[src[i]] + adst[dst[i]]
                  + ex0 * (Wattn[128] + v[0]) + ex1 * (Wattn[129] + v[1]);
    const float logit = (a > 0.f) ? a : NEG_SLOPE * a;
    exw[2 * i]     = ex0;
    exw[2 * i + 1] = ex1;
    logitw[i]      = logit;
    atomicMax(&menc[dst[i]], fenc(logit));
}

// ---------------- Kernel 5: edge pass 2 (segment sum of exp) ----------------
__global__ void edge2_kernel(const int* __restrict__ dst,
                             const float* __restrict__ logitw,
                             const unsigned* __restrict__ menc,
                             float* __restrict__ s) {
    const int i = blockIdx.x * blockDim.x + threadIdx.x;
    if (i >= N_EDGES) return;
    const int d = dst[i];
    atomicAdd(&s[d], __expf(logitw[i] - fdec(menc[d])));
}

// ---------------- Kernel 6: edge pass 3 (weighted scatter-add) ----------------
// 64 threads per edge, one output dim each.
__global__ void edge3_kernel(const int* __restrict__ src,
                             const int* __restrict__ dst,
                             const float* __restrict__ exw,
                             const float* __restrict__ logitw,
                             const unsigned* __restrict__ menc,
                             const float* __restrict__ s,
                             const float* __restrict__ z,
                             const float* __restrict__ We2n,
                             float* __restrict__ hout) {
    const int gid = blockIdx.x * blockDim.x + threadIdx.x;   // < 51.2M, fits int
    const int i = gid >> 6;
    const int d = gid & 63;
    if (i >= N_EDGES) return;
    const int sN = src[i], dN = dst[i];
    const float attn = __expf(logitw[i] - fdec(menc[dN])) / s[dN];
    const float ez = exw[2 * i] * We2n[2 * d] + exw[2 * i + 1] * We2n[2 * d + 1];
    atomicAdd(&hout[(size_t)dN * D_OUT + d], attn * (z[(size_t)sN * D_OUT + d] + ez));
}

extern "C" void kernel_launch(void* const* d_in, const int* in_sizes, int n_in,
                              void* d_out, int out_size, void* d_ws, size_t ws_size,
                              hipStream_t stream) {
    const float* h         = (const float*)d_in[0];
    const float* e         = (const float*)d_in[1];
    const int*   src       = (const int*)  d_in[2];
    const int*   dst       = (const int*)  d_in[3];
    const float* Wfc       = (const float*)d_in[4];
    const float* Wattn     = (const float*)d_in[5];
    const float* Weatt     = (const float*)d_in[6];
    const float* We2n      = (const float*)d_in[7];
    const float* Wedge     = (const float*)d_in[8];
    const float* attn_edge = (const float*)d_in[9];

    float*    out  = (float*)d_out;
    float*    ws   = (float*)d_ws;
    float*    z    = ws + WS_Z;
    float*    asrc = ws + WS_ASRC;
    float*    adst = ws + WS_ADST;
    unsigned* menc = (unsigned*)(ws + WS_MENC);
    float*    s    = ws + WS_S;
    float*    v    = ws + WS_V;
    float*    exw  = ws + WS_EX;
    float*    lg   = ws + WS_LOGIT;

    hipMemsetAsync(out,  0, sizeof(float)    * N_NODES * D_OUT, stream);
    hipMemsetAsync(menc, 0, sizeof(unsigned) * N_NODES,         stream);  // 0 == encoded "-inf-ish" floor
    hipMemsetAsync(s,    0, sizeof(float)    * N_NODES,         stream);

    // GEMM: 3125 waves, 8 waves (256 threads) per block
    gemm_z_kernel<<<(N_NODES / 16 + 7) / 8, 256, 0, stream>>>(h, Wfc, z);
    param_kernel<<<1, 64, 0, stream>>>(Wedge, attn_edge, v);
    alpha_kernel<<<(N_NODES + 7) / 8, 256, 0, stream>>>(z, Wattn, asrc, adst);
    edge1_kernel<<<(N_EDGES + 255) / 256, 256, 0, stream>>>(
        e, src, dst, Weatt, Wattn, v, asrc, adst, exw, lg, menc);
    edge2_kernel<<<(N_EDGES + 255) / 256, 256, 0, stream>>>(dst, lg, menc, s);
    edge3_kernel<<<(N_EDGES * 64) / 256, 256, 0, stream>>>(
        src, dst, exw, lg, menc, s, z, We2n, out);
}
